// IrregularModel_42004780155212
// MI455X (gfx1250) — matrix operand order repaired and checked by reading, not compiled, not run on hardware
//
#include <hip/hip_runtime.h>
#include <math.h>

typedef __attribute__((ext_vector_type(16))) _Float16 v16h;
typedef __attribute__((ext_vector_type(8)))  _Float16 v8h;
typedef __attribute__((ext_vector_type(8)))  float    v8f;
typedef _Float16 f16;

#define CONST_ static constexpr int
CONST_ Bc = 8, Lc = 336, Nc = 12, Dc = 64, Tc = 64, Fc = 256, LPc = 96, Hh = 8, DHc = 8;
CONST_ BL  = Bc * Lc;          // 2688
CONST_ Mr  = Tc * BL;          // 172032  rows of flattened [T,B,L]
CONST_ BDL = Bc * Dc * Lc;     // 172032  per-t slab of conv output
static constexpr float EPSb = 1e-5f;

// ---------------- input projection: x[b,l,d] = X[b,l,:] . w_in[d,:] + b_in[d] -----------
__global__ void k_inproj(const float* __restrict__ X, const float* __restrict__ w,
                         const float* __restrict__ b, float* __restrict__ xo) {
  int tid = blockIdx.x * blockDim.x + threadIdx.x;
  if (tid >= BL * Dc) return;
  int d = tid % Dc, bl = tid / Dc;
  const float* xr = X + (size_t)bl * Nc;
  const float* wr = w + (size_t)d * Nc;
  float a = b[d];
#pragma unroll
  for (int n = 0; n < Nc; ++n) a += xr[n] * wr[n];
  xo[tid] = a;
}

// ---------------- conv(1,3) over L, channel-major layout [t][b][d][l] (f16) -------------
__global__ void k_conv(const float* __restrict__ x, const float* __restrict__ cw,
                       const float* __restrict__ cb, f16* __restrict__ hc) {
  int tid = blockIdx.x * blockDim.x + threadIdx.x;
  if (tid >= Tc * BDL) return;
  int l = tid % Lc;
  int d = (tid / Lc) % Dc;
  int b = (tid / (Lc * Dc)) % Bc;
  int t = tid / (Lc * Dc * Bc);
  float a = cb[t];
#pragma unroll
  for (int j = 0; j < 3; ++j) {
    int ll = l + j - 1;
    if (ll >= 0 && ll < Lc) a += cw[t * 3 + j] * x[((size_t)(b * Lc + ll)) * Dc + d];
  }
  hc[tid] = (f16)a;  // tid == t*BDL + (b*Dc+d)*Lc + l
}

// ---------------- BN stats, channel-major (one block per channel, contiguous slab) ------
__global__ void k_stats_major(const f16* __restrict__ y, float* __restrict__ mean,
                              float* __restrict__ rstd, int chunk) {
  __shared__ float s1[256], s2[256];
  int c = blockIdx.x;
  const f16* p = y + (size_t)c * chunk;
  float a = 0.f, b = 0.f;
  for (int i = threadIdx.x; i < chunk; i += 256) {
    float v = (float)p[i];
    a += v; b += v * v;
  }
  s1[threadIdx.x] = a; s2[threadIdx.x] = b;
  __syncthreads();
  for (int o = 128; o > 0; o >>= 1) {
    if ((int)threadIdx.x < o) { s1[threadIdx.x] += s1[threadIdx.x + o]; s2[threadIdx.x] += s2[threadIdx.x + o]; }
    __syncthreads();
  }
  if (threadIdx.x == 0) {
    float m = s1[0] / (float)chunk;
    float v = s2[0] / (float)chunk - m * m;
    mean[c] = m; rstd[c] = rsqrtf(v + EPSb);
  }
}

// ---------------- coalesced partial BN stats for [rows, E] channel-minor tensors --------
// consecutive threads read consecutive f16; channel = tid % E is loop-invariant (256 % E == 0)
__global__ void k_zero(float* __restrict__ p, int n) {
  int t = blockIdx.x * 256 + threadIdx.x;
  if (t < n) p[t] = 0.f;
}

__global__ void k_stats_part(const f16* __restrict__ y, float* __restrict__ sum,
                             float* __restrict__ sumsq, int E, int rows) {
  __shared__ float ls[256], ls2[256];
  const int rowsPer = rows / gridDim.x;
  const size_t base = (size_t)blockIdx.x * rowsPer * E;
  const int total = rowsPer * E;
  float a = 0.f, b = 0.f;
  for (int i = threadIdx.x; i < total; i += 256) {
    float v = (float)y[base + i];
    a += v; b += v * v;
  }
  ls[threadIdx.x] = a; ls2[threadIdx.x] = b;
  __syncthreads();
  if ((int)threadIdx.x < E) {
    float sa = 0.f, sb = 0.f;
    for (int j = threadIdx.x; j < 256; j += E) { sa += ls[j]; sb += ls2[j]; }
    atomicAdd(&sum[threadIdx.x], sa);     // global_atomic_add_f32
    atomicAdd(&sumsq[threadIdx.x], sb);
  }
}

__global__ void k_stats_fin(const float* __restrict__ sum, const float* __restrict__ sumsq,
                            float* __restrict__ mean, float* __restrict__ rstd,
                            int E, float invn) {
  int c = threadIdx.x;
  if (c >= E) return;
  float m = sum[c] * invn;
  float v = sumsq[c] * invn - m * m;
  mean[c] = m; rstd[c] = rsqrtf(v + EPSb);
}

// ------- encoder: BN(channel=t) + LIF + LIF, transpose [t,b,d,l] -> [t,b,l,d] spikes ----
__global__ void k_bnlif_conv(const f16* __restrict__ hc, const float* __restrict__ mean,
                             const float* __restrict__ rstd, const float* __restrict__ g,
                             const float* __restrict__ be, f16* __restrict__ s) {
  int tid = blockIdx.x * blockDim.x + threadIdx.x;
  if (tid >= BDL) return;
  int l = tid % Lc;
  int d = (tid / Lc) % Dc;
  int b = tid / (Lc * Dc);
  size_t base = (size_t)(b * Dc + d) * Lc + l;
  float v1 = 0.f, v2 = 0.f;
  for (int t = 0; t < Tc; ++t) {
    float x = (float)hc[(size_t)t * BDL + base];
    x = (x - mean[t]) * rstd[t] * g[t] + be[t];
    v1 = 0.5f * (v1 + x);
    float s1 = (v1 > 1.0f) ? 1.0f : 0.0f;
    v1 *= (1.0f - s1);
    v2 = 0.5f * (v2 + s1);
    float s2v = (v2 > 1.0f) ? 1.0f : 0.0f;
    v2 *= (1.0f - s2v);
    s[((size_t)(t * Bc + b) * Lc + l) * Dc + d] = (f16)s2v;
  }
}

// ------- weight convert: W[E,K] f32 -> packed f16 Wt[(K/16)][E][16] (WMMA-B friendly) ---
__global__ void k_cvt(const float* __restrict__ W, f16* __restrict__ Wt, int E, int K) {
  int tid = blockIdx.x * blockDim.x + threadIdx.x;
  if (tid >= E * K) return;
  int e = tid / K, k = tid % K;
  Wt[((size_t)(k >> 4) * E + e) * 16 + (k & 15)] = (f16)W[tid];
}

// ---------------- WMMA GEMM: C[Mr,N](f16) = A[Mr,K](f16) x Wpacked (+bias) --------------
// block = 256 thr = 8 waves; wave w owns m-strip (blockIdx.x*8+w)*16, 4 n-tiles of group
// blockIdx.y*64. A fragment loaded once per k-step (2x b128); all 4 B fragments are
// loaded up-front so the 4 WMMAs issue back-to-back (XDL pipelining).
__global__ void k_gemm(const f16* __restrict__ A, const f16* __restrict__ Bt,
                       const float* __restrict__ bias, f16* __restrict__ C,
                       int K, int N) {
  const int lane = threadIdx.x & 31;
  const int wave = threadIdx.x >> 5;
  const int m0 = (blockIdx.x * 8 + wave) * 16;
  const int ng = blockIdx.y * 64;
  const int mr = lane & 15;
  const bool hi = lane >= 16;
  const int nB = ng + (lane & 15);
  const f16* Arow = A + (size_t)(m0 + mr) * K;
  v8f acc[4] = {v8f{}, v8f{}, v8f{}, v8f{}};
  for (int kk = 0; kk < K; kk += 32) {
    if (kk + 32 < K) __builtin_prefetch(Arow + kk + 32, 0, 1);  // global_prefetch_b8
    // A 16x32 fragment (contiguous k, as in B): lanes 0-15 K in [kk,kk+16), lanes 16-31 K in [kk+16,kk+32)
    const int ab = kk + (hi ? 16 : 0);
    v8h a0 = *(const v8h*)(Arow + ab);        // global_load_b128
    v8h a1 = *(const v8h*)(Arow + ab + 8);    // global_load_b128
    v16h af = __builtin_shufflevector(a0, a1, 0, 1, 2, 3, 4, 5, 6, 7,
                                      8, 9, 10, 11, 12, 13, 14, 15);
    // B 32x16 fragments: 16 consecutive K contiguous per column in packed Wt.
    // Load all four n-tiles first, then issue the WMMAs back-to-back.
    const int kb = kk + (hi ? 16 : 0);
    const f16* bp = Bt + ((size_t)(kb >> 4) * N + nB) * 16;
    v16h bf[4];
#pragma unroll
    for (int j = 0; j < 4; ++j) {
      const f16* bpj = bp + (size_t)j * 256;  // next 16 columns
      v8h b0 = *(const v8h*)bpj;
      v8h b1 = *(const v8h*)(bpj + 8);
      bf[j] = __builtin_shufflevector(b0, b1, 0, 1, 2, 3, 4, 5, 6, 7,
                                      8, 9, 10, 11, 12, 13, 14, 15);
    }
#pragma unroll
    for (int j = 0; j < 4; ++j)
      acc[j] = __builtin_amdgcn_wmma_f32_16x16x32_f16(false, af, false, bf[j],
                                                      (short)0, acc[j], false, false);
  }
  // C layout: VGPR r -> M = r (+8 for lanes 16-31), N = lane&15
#pragma unroll
  for (int j = 0; j < 4; ++j) {
    const int cn = nB + j * 16;
    const float bb = bias ? bias[cn] : 0.0f;
#pragma unroll
    for (int r = 0; r < 8; ++r) {
      int cm = m0 + r + (hi ? 8 : 0);
      C[(size_t)cm * N + cn] = (f16)(acc[j][r] + bb);
    }
  }
}

// -------- BN(channel-minor, optional) + LIF over t + optional residual add --------------
__global__ void k_bnlif(const f16* __restrict__ Y, const float* __restrict__ mean,
                        const float* __restrict__ rstd, const float* __restrict__ g,
                        const float* __restrict__ be, const f16* __restrict__ resid,
                        f16* __restrict__ S, int E, float vth, int use_bn) {
  int tid = blockIdx.x * blockDim.x + threadIdx.x;
  if (tid >= BL * E) return;
  int e = tid % E;
  float m = 0.f, rs = 1.f, gg = 1.f, bb = 0.f;
  if (use_bn) { m = mean[e]; rs = rstd[e]; gg = g[e]; bb = be[e]; }
  const size_t stride = (size_t)BL * E;
  float v = 0.f;
  for (int t = 0; t < Tc; ++t) {
    size_t idx = (size_t)t * stride + tid;
    float x = (float)Y[idx];
    if (use_bn) x = (x - m) * rs * gg + bb;
    v = 0.5f * (v + x);
    float sp = (v > vth) ? 1.0f : 0.0f;
    v *= (1.0f - sp);
    float o = sp;
    if (resid) o += (float)resid[idx];
    S[idx] = (f16)o;
  }
}

// ---------------- kv[t,b,h,d,e] = sum_l k[t,b,l,h,d] * v[t,b,l,h,e] ---------------------
__global__ void k_kv(const f16* __restrict__ ks, const f16* __restrict__ vs,
                     float* __restrict__ kv) {
  int blk = blockIdx.x;            // (t*Bc+b)*Hh + h
  int h = blk % Hh;
  int tb = blk / Hh;               // t*Bc + b
  int d = threadIdx.x >> 3, e = threadIdx.x & 7;
  const f16* kp = ks + ((size_t)tb * Lc) * Dc + h * DHc + d;
  const f16* vp = vs + ((size_t)tb * Lc) * Dc + h * DHc + e;
  float a = 0.f;
  for (int l = 0; l < Lc; ++l)
    a += (float)kp[(size_t)l * Dc] * (float)vp[(size_t)l * Dc];
  kv[(size_t)blk * 64 + threadIdx.x] = a;   // threadIdx.x == d*8+e
}

// ---------------- y[t,b,l,h,e] = 0.125 * sum_d q[t,b,l,h,d] * kv[t,b,h,d,e] -------------
__global__ void k_attn_y(const f16* __restrict__ qs, const float* __restrict__ kv,
                         f16* __restrict__ y) {
  int tid = blockIdx.x * blockDim.x + threadIdx.x;
  if (tid >= Mr * Dc) return;
  int c = tid % Dc;
  size_t row = (size_t)tid / Dc;
  int h = c >> 3, e = c & 7;
  int tb = (int)(row / Lc);
  const f16* qp = qs + row * (size_t)Dc + h * DHc;
  const float* kp = kv + ((size_t)tb * Hh + h) * 64 + e;
  float a = 0.f;
#pragma unroll
  for (int d = 0; d < 8; ++d) a += (float)qp[d] * kp[d * 8];
  y[tid] = (f16)(0.125f * a);
}

// ---------------- mean over t: mb[b,l,d] = mean_t s[t,b,l,d] ----------------------------
__global__ void k_mean(const f16* __restrict__ s, float* __restrict__ mb) {
  int tid = blockIdx.x * blockDim.x + threadIdx.x;
  if (tid >= BL * Dc) return;
  const size_t stride = (size_t)BL * Dc;
  float a = 0.f;
  for (int t = 0; t < Tc; ++t) a += (float)s[(size_t)t * stride + tid];
  mb[tid] = a * (1.0f / (float)Tc);
}

// ---------------- head: out[b,p,n] = mb[b,240+p,:] . w_out[n,:] + b_out[n] --------------
__global__ void k_out(const float* __restrict__ mb, const float* __restrict__ wo,
                      const float* __restrict__ bo, float* __restrict__ out) {
  int tid = blockIdx.x * blockDim.x + threadIdx.x;
  if (tid >= Bc * LPc * Nc) return;
  int n = tid % Nc;
  int p = (tid / Nc) % LPc;
  int b = tid / (Nc * LPc);
  int l = Lc - LPc + p;
  const float* mr = mb + (size_t)(b * Lc + l) * Dc;
  const float* wr = wo + (size_t)n * Dc;
  float a = bo[n];
#pragma unroll 8
  for (int d = 0; d < Dc; ++d) a += mr[d] * wr[d];
  out[tid] = a;
}

// ========================================================================================
extern "C" void kernel_launch(void* const* d_in, const int* in_sizes, int n_in,
                              void* d_out, int out_size, void* d_ws, size_t ws_size,
                              hipStream_t stream) {
  (void)in_sizes; (void)n_in; (void)out_size; (void)ws_size;
  // input order: X, tp_true, tp_pred, w_in, b_in, conv_w, conv_b, g_e, b_e, w_out, b_out,
  //              then 2 blocks x {wq,gq,bq,wk,gk,bk,wv,gv,bv,wp,bp,gp,bpb,
  //                               w1,b1,g1,bb1,w2,b2,g2,bb2}
  const float* X      = (const float*)d_in[0];
  const float* w_in   = (const float*)d_in[3];
  const float* b_in   = (const float*)d_in[4];
  const float* conv_w = (const float*)d_in[5];
  const float* conv_b = (const float*)d_in[6];
  const float* g_e    = (const float*)d_in[7];
  const float* b_e    = (const float*)d_in[8];
  const float* w_out  = (const float*)d_in[9];
  const float* b_out  = (const float*)d_in[10];

  char* ws = (char*)d_ws;
  const size_t SLAB = (size_t)Mr * Dc * sizeof(f16);        // 22,020,096 B
  float* xb    = (float*)(ws + 0);                          // [BL,Dc] f32, later mean buf
  float* meanA = (float*)(ws + (1ull << 20));               // 256 f32
  float* rstdA = meanA + 256;
  float* sumA  = meanA + 512;                               // sum + sumsq (2*256 f32)
  f16*   wslot = (f16*)(ws + (1ull << 20) + 8192);          // up to 16384 f16 packed
  float* kvb   = (float*)(ws + (2ull << 20));               // [Tc*Bc*Hh, 64] f32 = 1 MB
  f16*   sbuf  = (f16*)(ws + (4ull << 20));                 // s spikes [T,B,L,D] f16
  f16*   spk   = (f16*)(ws + (28ull << 20));                // q/k/v or mlp-hidden spikes
  f16*   bufA  = (f16*)(ws + (124ull << 20));               // GEMM out / conv pre-act
  f16* hc   = bufA;                                         // alias (consumed before GEMMs)
  f16* qs   = spk;
  f16* ksb  = (f16*)((char*)spk + SLAB);
  f16* vsb  = (f16*)((char*)spk + 2 * SLAB);
  f16* atts = ksb;                                          // reuse k slab after kv einsum
  f16* hspk = spk;                                          // reuse whole region in MLP

  auto nb = [](int n) { return dim3((unsigned)((n + 255) / 256)); };
  auto stats = [&](const f16* y, int E) {
    k_zero<<<2, 256, 0, stream>>>(sumA, 512);
    k_stats_part<<<256, 256, 0, stream>>>(y, sumA, sumA + 256, E, Mr);
    k_stats_fin<<<1, 256, 0, stream>>>(sumA, sumA + 256, meanA, rstdA, E, 1.0f / (float)Mr);
  };

  // encoder
  k_inproj<<<nb(BL * Dc), 256, 0, stream>>>(X, w_in, b_in, xb);
  k_conv<<<nb(Tc * BDL), 256, 0, stream>>>(xb, conv_w, conv_b, hc);
  k_stats_major<<<Tc, 256, 0, stream>>>(hc, meanA, rstdA, BDL);
  k_bnlif_conv<<<nb(BDL), 256, 0, stream>>>(hc, meanA, rstdA, g_e, b_e, sbuf);

  for (int blk = 0; blk < 2; ++blk) {
    int base = 11 + 21 * blk;
    const float* wq  = (const float*)d_in[base + 0];
    const float* gq  = (const float*)d_in[base + 1];
    const float* bq  = (const float*)d_in[base + 2];
    const float* wk  = (const float*)d_in[base + 3];
    const float* gk  = (const float*)d_in[base + 4];
    const float* bk  = (const float*)d_in[base + 5];
    const float* wv  = (const float*)d_in[base + 6];
    const float* gv  = (const float*)d_in[base + 7];
    const float* bv  = (const float*)d_in[base + 8];
    const float* wp  = (const float*)d_in[base + 9];
    const float* bp  = (const float*)d_in[base + 10];
    const float* gp  = (const float*)d_in[base + 11];
    const float* bpb = (const float*)d_in[base + 12];
    const float* w1  = (const float*)d_in[base + 13];
    const float* b1  = (const float*)d_in[base + 14];
    const float* g1w = (const float*)d_in[base + 15];
    const float* bb1 = (const float*)d_in[base + 16];
    const float* w2  = (const float*)d_in[base + 17];
    const float* b2  = (const float*)d_in[base + 18];
    const float* g2w = (const float*)d_in[base + 19];
    const float* bb2 = (const float*)d_in[base + 20];

    const dim3 gemm64(Mr / 128, Dc / 64);   // N = 64
    const dim3 gemm256(Mr / 128, Fc / 64);  // N = 256

    // q / k / v projections -> BN -> LIF spikes
    const float* wqkv[3] = {wq, wk, wv};
    const float* gqkv[3] = {gq, gk, gv};
    const float* bqkv[3] = {bq, bk, bv};
    f16* sqkv[3] = {qs, ksb, vsb};
    for (int i = 0; i < 3; ++i) {
      k_cvt<<<nb(Dc * Dc), 256, 0, stream>>>(wqkv[i], wslot, Dc, Dc);
      k_gemm<<<gemm64, 256, 0, stream>>>(sbuf, wslot, nullptr, bufA, Dc, Dc);
      stats(bufA, Dc);
      k_bnlif<<<nb(BL * Dc), 256, 0, stream>>>(bufA, meanA, rstdA, gqkv[i], bqkv[i],
                                               nullptr, sqkv[i], Dc, 1.0f, 1);
    }
    // attention via associativity: kv = k^T v; y = 0.125 * q kv; attn LIF (vth=0.5)
    k_kv<<<Tc * Bc * Hh, 64, 0, stream>>>(ksb, vsb, kvb);
    k_attn_y<<<nb(Mr * Dc), 256, 0, stream>>>(qs, kvb, bufA);
    k_bnlif<<<nb(BL * Dc), 256, 0, stream>>>(bufA, nullptr, nullptr, nullptr, nullptr,
                                             nullptr, atts, Dc, 0.5f, 0);
    // proj -> BN -> LIF -> residual add into s (in place)
    k_cvt<<<nb(Dc * Dc), 256, 0, stream>>>(wp, wslot, Dc, Dc);
    k_gemm<<<gemm64, 256, 0, stream>>>(atts, wslot, bp, bufA, Dc, Dc);
    stats(bufA, Dc);
    k_bnlif<<<nb(BL * Dc), 256, 0, stream>>>(bufA, meanA, rstdA, gp, bpb,
                                             sbuf, sbuf, Dc, 1.0f, 1);
    // MLP: D -> DFF -> BN -> LIF -> DFF -> D -> BN -> LIF -> residual
    k_cvt<<<nb(Fc * Dc), 256, 0, stream>>>(w1, wslot, Fc, Dc);
    k_gemm<<<gemm256, 256, 0, stream>>>(sbuf, wslot, b1, bufA, Dc, Fc);
    stats(bufA, Fc);
    k_bnlif<<<nb(BL * Fc), 256, 0, stream>>>(bufA, meanA, rstdA, g1w, bb1,
                                             nullptr, hspk, Fc, 1.0f, 1);
    k_cvt<<<nb(Dc * Fc), 256, 0, stream>>>(w2, wslot, Dc, Fc);
    k_gemm<<<gemm64, 256, 0, stream>>>(hspk, wslot, b2, bufA, Fc, Dc);
    stats(bufA, Dc);
    k_bnlif<<<nb(BL * Dc), 256, 0, stream>>>(bufA, meanA, rstdA, g2w, bb2,
                                             sbuf, sbuf, Dc, 1.0f, 1);
  }

  // head: mean over T, output projection, slice last LP positions
  k_mean<<<nb(BL * Dc), 256, 0, stream>>>(sbuf, xb);
  k_out<<<nb(Bc * LPc * Nc), 256, 0, stream>>>(xb, w_out, b_out, (float*)d_out);
}
